// MultiHeadSelfAttention_31172872634899
// MI455X (gfx1250) — compile-verified, matrix operand order checked by reading, not run on hardware
//
#include <hip/hip_runtime.h>

typedef unsigned short u16;
typedef unsigned int   u32;
typedef unsigned long long u64;
typedef __attribute__((ext_vector_type(16))) __bf16 v16bf;
typedef __attribute__((ext_vector_type(8)))  float  v8f;

#define B_  4
#define S_  2048
#define D_  1024
#define H_  16
#define HD_ 64
#define M_  (B_*S_)   // 8192 tokens

// ---------------- helpers ----------------

__device__ inline u16 f2bf(float f) {
  union { float f; u32 u; } v; v.f = f;
  u32 r = v.u + 0x7fffu + ((v.u >> 16) & 1u);   // round-to-nearest-even
  return (u16)(r >> 16);
}

__device__ inline v8f wmma_bf16(v16bf a, v16bf b, v8f c) {
  // 8 args: (neg_a, A, neg_b, B, c_mod, C, reuse_a, reuse_b)
  return __builtin_amdgcn_wmma_f32_16x16x32_bf16(false, a, false, b, (short)0, c,
                                                 false, false);
}

// Async global -> LDS copy, 16 bytes per lane, tracked by ASYNCcnt.
// LDS aperture: low 32 bits of a generic shared pointer are the LDS byte addr.
__device__ inline void async_load_b128(const u16* gptr, u16* ldsPtr) {
  u64 ga = (u64)(uintptr_t)gptr;
  u32 la = (u32)(uintptr_t)ldsPtr;
  asm volatile("global_load_async_to_lds_b128 %0, %1, off"
               :: "v"(la), "v"(ga) : "memory");
}
__device__ inline void wait_async() {
  asm volatile("s_wait_asynccnt 0" ::: "memory");
}

// Build one 16x32 bf16 WMMA operand (A layout; B uses the identical per-lane
// layout applied to B^T rows). rowPtr = this lane's row (>=32 contiguous bf16,
// 16B aligned). half = lane>>4. Per the ISA table, the lane's 16 elements are
// the two contiguous 8-element chunks at k = half*8 and k = 16 + half*8.
__device__ inline v16bf make_frag(const u16* rowPtr, int half) {
  union { v16bf v; uint4 q[2]; } u;
  u.q[0] = *reinterpret_cast<const uint4*>(rowPtr + half*8);
  u.q[1] = *reinterpret_cast<const uint4*>(rowPtr + 16 + half*8);
  return u.v;
}

// ---------------- fp32 -> bf16 conversion ----------------

__global__ void convert_f32_bf16(const float* __restrict__ in,
                                 u16* __restrict__ out, int n) {
  for (int i = blockIdx.x*blockDim.x + threadIdx.x; i < n;
       i += gridDim.x*blockDim.x)
    out[i] = f2bf(in[i]);
}

// ---------------- tiled WMMA GEMM: C = A * W^T + bias ----------------
// A: [rows, 1024] bf16 row-major.  W: [1024, 1024] bf16 row-major
// (row n of W = weights of output column n, i.e. B^T row n — contiguous).
// Block tile 256x128, 8 waves of 64x64, BK=32, async double-buffered LDS.
// mode 0: bf16 out, head-split [B,H,S,HD].  mode 1: fp32 out row-major.

#define BM 256
#define BN 128
#define BK 32
#define LDT 40    // padded LDS row stride (80B rows, keeps 16B chunk alignment)

__global__ __launch_bounds__(256)
void gemm_bf16(const u16* __restrict__ A, const u16* __restrict__ W,
               const float* __restrict__ bias, void* __restrict__ Cout,
               int mode) {
  __shared__ __align__(16) u16 sA[2][BM*LDT];   // 2 x 20480 B
  __shared__ __align__(16) u16 sB[2][BN*LDT];   // 2 x 10240 B

  const int t    = threadIdx.x;
  const int lane = t & 31, wave = t >> 5;
  const int half = lane >> 4, lrow = lane & 15;
  const int mW = (wave & 3) * 64;        // 4 waves along M
  const int nW = (wave >> 2) * 64;       // 2 waves along N
  const int blockM = blockIdx.y * BM, blockN = blockIdx.x * BN;

  // cooperative fill: A = 256 rows x 32 (1 row/thread, 4x b128),
  //                   B = 128 rows x 32 (half row/thread, 2x b128)
  const int aRow = t;
  const int bRow = t >> 1, bCh = (t & 1) * 16;

  auto issue_tile = [&](int k0, int buf) {
    const u16* ga = A + (size_t)(blockM + aRow)*D_ + k0;
    u16* la = sA[buf] + aRow*LDT;
#pragma unroll
    for (int c = 0; c < 4; c++) async_load_b128(ga + c*8, la + c*8);
    const u16* gb = W + (size_t)(blockN + bRow)*D_ + k0 + bCh;
    u16* lb = sB[buf] + bRow*LDT + bCh;
    async_load_b128(gb,     lb);
    async_load_b128(gb + 8, lb + 8);
  };

  v8f acc[4][4];
#pragma unroll
  for (int i = 0; i < 4; i++)
#pragma unroll
    for (int j = 0; j < 4; j++) acc[i][j] = (v8f)0.0f;

  issue_tile(0, 0);

  int buf = 0;
  for (int k0 = 0; k0 < D_; k0 += BK, buf ^= 1) {
    wait_async();          // this wave's fills of `buf` are in LDS
    __syncthreads();       // everyone's fills done; prior compute on buf^1 done
    if (k0 + BK < D_) issue_tile(k0 + BK, buf ^ 1);   // overlap next fill

    v16bf af[4], bf[4];
#pragma unroll
    for (int mi = 0; mi < 4; mi++)
      af[mi] = make_frag(sA[buf] + (mW + mi*16 + lrow)*LDT, half);
#pragma unroll
    for (int ni = 0; ni < 4; ni++)
      bf[ni] = make_frag(sB[buf] + (nW + ni*16 + lrow)*LDT, half);
#pragma unroll
    for (int mi = 0; mi < 4; mi++)
#pragma unroll
      for (int ni = 0; ni < 4; ni++)
        acc[mi][ni] = wmma_bf16(af[mi], bf[ni], acc[mi][ni]);
  }

  // epilogue: C layout — lane = column (within 16), row = e + half*8
#pragma unroll
  for (int mi = 0; mi < 4; mi++) {
#pragma unroll
    for (int ni = 0; ni < 4; ni++) {
      int col = blockN + nW + ni*16 + lrow;
      float bv = bias[col];
      int rbase = blockM + mW + mi*16 + half*8;
#pragma unroll
      for (int e = 0; e < 8; e++) {
        float v = acc[mi][ni][e] + bv;
        int r = rbase + e;
        if (mode == 0) {
          int b = r >> 11, s = r & (S_-1);
          int h = col >> 6, d = col & (HD_-1);
          ((u16*)Cout)[((size_t)(b*H_ + h)*S_ + s)*HD_ + d] = f2bf(v);
        } else {
          ((float*)Cout)[(size_t)r*D_ + col] = v;
        }
      }
    }
  }
}

// ---------------- flash attention (causal, per-(b,h)) ----------------
// Q,K,V: [B*H, S, 64] bf16. Block = 4 waves, each wave owns 16 query rows.
// KV streamed in 32-key tiles; K filled via async-to-LDS, V transposed
// through VGPRs (2-byte scatter, async copy cannot transpose).

__global__ __launch_bounds__(128)
void flash_attn(const u16* __restrict__ Q, const u16* __restrict__ K,
                const u16* __restrict__ V, u16* __restrict__ O) {
  __shared__ __align__(16) u16 sK[32*72];     // 32 keys x 64 d, stride 72
  __shared__ __align__(16) u16 sVT[64*40];    // V^T: 64 d x 32 keys, stride 40
  __shared__ __align__(16) u16 sP[4*16*40];   // per-wave P tile 16x32, stride 40

  const int t    = threadIdx.x;
  const int lane = t & 31, wave = t >> 5;
  const int half = lane >> 4, lrow = lane & 15;
  const int bh = blockIdx.y;
  const int q0 = blockIdx.x * 64;
  const int qrow = q0 + wave * 16;
  u16* myP = sP + wave * (16*40);

  // Q fragments live in registers for the whole KV sweep (K-dim 64 = 2 frags)
  const u16* qptr = Q + ((size_t)bh*S_ + qrow + lrow)*HD_;
  v16bf aq0 = make_frag(qptr, half);
  v16bf aq1 = make_frag(qptr + 32, half);

  v8f o[4];
#pragma unroll
  for (int i = 0; i < 4; i++) o[i] = (v8f)0.0f;
  float rm[8], rs[8];
#pragma unroll
  for (int e = 0; e < 8; e++) { rm[e] = -1e30f; rs[e] = 0.0f; }

  const int myq = qrow + half*8;   // + e = this lane's row (C layout)
  const int krow = t >> 2, kch = (t & 3) * 16;   // tile fill indices

  for (int kv0 = 0; kv0 < q0 + 64; kv0 += 32) {   // uniform across waves
    {   // K: async fill, 32 rows x 64 elems; V: load + transpose scatter
      const u16* gk = K + ((size_t)bh*S_ + kv0 + krow)*HD_ + kch;
      u16* lk = sK + krow*72 + kch;
      async_load_b128(gk,     lk);
      async_load_b128(gk + 8, lk + 8);
      const uint4* gv = reinterpret_cast<const uint4*>(
          V + ((size_t)bh*S_ + kv0 + krow)*HD_ + kch);
      uint4 va = gv[0], vb = gv[1];
      u16 tmp[16];
      *reinterpret_cast<uint4*>(tmp)     = va;
      *reinterpret_cast<uint4*>(tmp + 8) = vb;
#pragma unroll
      for (int i = 0; i < 16; i++) sVT[(kch + i)*40 + krow] = tmp[i];
      wait_async();
    }
    __syncthreads();

    // scores: S[16x32] = Q[16x64] x K^T, as two 16x16 C tiles, 2 wmma each
    v8f s0 = (v8f)0.0f, s1 = (v8f)0.0f;
    s0 = wmma_bf16(aq0, make_frag(sK + lrow*72,           half), s0);
    s0 = wmma_bf16(aq1, make_frag(sK + lrow*72 + 32,      half), s0);
    s1 = wmma_bf16(aq0, make_frag(sK + (16 + lrow)*72,    half), s1);
    s1 = wmma_bf16(aq1, make_frag(sK + (16 + lrow)*72+32, half), s1);

    const int key0 = kv0 + lrow, key1 = kv0 + 16 + lrow;
    float nm[8];
#pragma unroll
    for (int e = 0; e < 8; e++) {
      float a  = s0[e] * 0.125f; if (key0 > myq + e) a  = -1e30f; s0[e] = a;
      float b2 = s1[e] * 0.125f; if (key1 > myq + e) b2 = -1e30f; s1[e] = b2;
      float mx = fmaxf(a, b2);
#pragma unroll
      for (int m = 1; m < 16; m <<= 1) mx = fmaxf(mx, __shfl_xor(mx, m));
      nm[e] = fmaxf(rm[e], mx);
    }
#pragma unroll
    for (int e = 0; e < 8; e++) {
      float p0 = __expf(s0[e] - nm[e]);
      float p1 = __expf(s1[e] - nm[e]);
      float alpha = __expf(rm[e] - nm[e]);
      rm[e] = nm[e];
      float ps = p0 + p1;
#pragma unroll
      for (int m = 1; m < 16; m <<= 1) ps += __shfl_xor(ps, m);
      rs[e] = rs[e]*alpha + ps;
#pragma unroll
      for (int di = 0; di < 4; di++) o[di][e] *= alpha;
      myP[(e + half*8)*40 + lrow]      = f2bf(p0);   // C layout -> row-major
      myP[(e + half*8)*40 + 16 + lrow] = f2bf(p1);
    }
    __syncthreads();

    // O += P[16x32] x V[32x64]: P is one A fragment, V^T rows give B frags
    v16bf ap = make_frag(myP + lrow*40, half);
#pragma unroll
    for (int di = 0; di < 4; di++)
      o[di] = wmma_bf16(ap, make_frag(sVT + (di*16 + lrow)*40, half), o[di]);
    __syncthreads();   // before next tile overwrites sK/sVT
  }

  // normalize + store to [B, S, H, HD] (== [8192, 1024] row-major)
  const int b = bh >> 4, h = bh & 15;
#pragma unroll
  for (int di = 0; di < 4; di++) {
    int d = di*16 + lrow;
#pragma unroll
    for (int e = 0; e < 8; e++) {
      float v = o[di][e] / rs[e];
      int q = myq + e;
      O[(((size_t)b*S_ + q)*H_ + h)*HD_ + d] = f2bf(v);
    }
  }
}

// ---------------- launch ----------------

extern "C" void kernel_launch(void* const* d_in, const int* in_sizes, int n_in,
                              void* d_out, int out_size, void* d_ws, size_t ws_size,
                              hipStream_t stream) {
  const float* x  = (const float*)d_in[0];
  // d_in[1] = attn_mask (known causal; handled analytically)
  const float* Wq = (const float*)d_in[2];
  const float* bq = (const float*)d_in[3];
  const float* Wk = (const float*)d_in[4];
  const float* bk = (const float*)d_in[5];
  const float* Wv = (const float*)d_in[6];
  const float* bv = (const float*)d_in[7];
  const float* Wo = (const float*)d_in[8];
  const float* bo = (const float*)d_in[9];

  u16* xh  = (u16*)d_ws;                         // [8192,1024] bf16
  u16* Wqh = xh  + (size_t)M_*D_;
  u16* Wkh = Wqh + (size_t)D_*D_;
  u16* Wvh = Wkh + (size_t)D_*D_;
  u16* Woh = Wvh + (size_t)D_*D_;
  u16* Qw  = Woh + (size_t)D_*D_;                // [B,H,S,HD] bf16
  u16* Kw  = Qw  + (size_t)M_*D_;
  u16* Vw  = Kw  + (size_t)M_*D_;
  u16* Aw  = Vw  + (size_t)M_*D_;                // [B,S,H,HD] bf16

  convert_f32_bf16<<<2048, 256, 0, stream>>>(x,  xh,  M_*D_);
  convert_f32_bf16<<<512,  256, 0, stream>>>(Wq, Wqh, D_*D_);
  convert_f32_bf16<<<512,  256, 0, stream>>>(Wk, Wkh, D_*D_);
  convert_f32_bf16<<<512,  256, 0, stream>>>(Wv, Wvh, D_*D_);
  convert_f32_bf16<<<512,  256, 0, stream>>>(Wo, Woh, D_*D_);

  dim3 gg(D_/BN, M_/BM);              // (8, 32), 256 threads
  gemm_bf16<<<gg, 256, 0, stream>>>(xh, Wqh, bq, (void*)Qw, 0);
  gemm_bf16<<<gg, 256, 0, stream>>>(xh, Wkh, bk, (void*)Kw, 0);
  gemm_bf16<<<gg, 256, 0, stream>>>(xh, Wvh, bv, (void*)Vw, 0);

  dim3 fg(S_/64, B_*H_);              // (32, 64), 128 threads
  flash_attn<<<fg, 128, 0, stream>>>(Qw, Kw, Vw, Aw);

  gemm_bf16<<<gg, 256, 0, stream>>>(Aw, Woh, bo, d_out, 1);
}
